// Transformer_5970004542178
// MI455X (gfx1250) — matrix operand order repaired and checked by reading, not compiled, not run on hardware
//
#include <hip/hip_runtime.h>
#include <math.h>

// ---------------------------------------------------------------------------
// Transformer block for MI455X (gfx1250, wave32, WMMA).
// All GEMMs use v_wmma_f32_16x16x32_f16 (f16 in, f32 accumulate).
// Needs ~160 MB of workspace (d_ws).
// ---------------------------------------------------------------------------

typedef _Float16 half_t;
typedef __attribute__((ext_vector_type(16))) _Float16 v16h;
typedef __attribute__((ext_vector_type(8)))  float    v8f;

#define Bn   32
#define Sn   512
#define Dn   512
#define Hn   8
#define DHn  64
#define DFFn 2048
#define NEGF (-4294967295.0f)
#define LNEPS 1e-3f

__device__ inline v8f wmma_f16(v16h a, v16h b, v8f c) {
  // (neg_a, A, neg_b, B, c_mod, C, reuse_a, reuse_b)
  return __builtin_amdgcn_wmma_f32_16x16x32_f16(false, a, false, b, (short)0, c,
                                                false, false);
}

// A-operand (16x32 f16): lane&15 = row M; lanes 0-15 hold K {0..7,16..23},
// lanes 16-31 hold K {8..15,24..31}.
__device__ inline v16h lds_load_A(const half_t As[128][40], int mbase, int lane) {
  int m  = mbase + (lane & 15);
  int kh = (lane >> 4) * 8;
  v16h a;
#pragma unroll
  for (int e = 0; e < 8; ++e) {
    a[e]     = As[m][kh + e];
    a[8 + e] = As[m][16 + kh + e];
  }
  return a;
}

// B-operand (32x16 f16): lane&15 = column N; lanes 0-15 hold K {0..7,16..23},
// lanes 16-31 hold K {8..15,24..31}.
__device__ inline v16h lds_load_B(const half_t Bs[32][136], int nbase, int lane) {
  int n  = nbase + (lane & 15);
  int kb = (lane >> 4) * 8;
  v16h b;
#pragma unroll
  for (int e = 0; e < 16; ++e) b[e] = Bs[kb + e + (e >= 8 ? 8 : 0)][n];
  return b;
}

// ---------------------------------------------------------------------------
// Generic GEMM: out = act(A[M,K] @ W[K,N]) (+ resid).  A is f16 or f32,
// W/resid are f32.  Writes f32 and/or f16 outputs.  M,N,K multiples of 128/32.
// Block tile 128x128, 256 threads = 8 waves, each wave 32x64.
// ---------------------------------------------------------------------------
template <bool A_F16, bool RELU>
__global__ __launch_bounds__(256) void gemm_wmma_kernel(
    const void* __restrict__ Aptr, const float* __restrict__ W,
    const float* __restrict__ resid, float* __restrict__ outF32,
    half_t* __restrict__ outF16, int M, int N, int K) {
  __shared__ half_t As[128][40];
  __shared__ half_t Bs[32][136];

  const int tid  = threadIdx.x;
  const int lane = tid & 31;
  const int w    = tid >> 5;
  const int wm   = (w >> 1) * 32;   // wave row origin in tile
  const int wn   = (w & 1) * 64;    // wave col origin in tile
  const int m0   = blockIdx.y * 128;
  const int n0   = blockIdx.x * 128;
  const int lh   = lane >> 4;
  const int ln   = lane & 15;

  v8f acc[2][4];
#pragma unroll
  for (int i = 0; i < 2; ++i)
#pragma unroll
    for (int j = 0; j < 4; ++j) {
      v8f z = {0.f, 0.f, 0.f, 0.f, 0.f, 0.f, 0.f, 0.f};
      acc[i][j] = z;
    }

  for (int k0 = 0; k0 < K; k0 += 32) {
    // Stage A tile (128x32) -> f16 LDS
    {
      int r = tid >> 1;
      int c = (tid & 1) * 16;
      if (A_F16) {
        const half_t* src = (const half_t*)Aptr + (size_t)(m0 + r) * K + k0 + c;
#pragma unroll
        for (int e = 0; e < 16; ++e) As[r][c + e] = src[e];
      } else {
        const float* src = (const float*)Aptr + (size_t)(m0 + r) * K + k0 + c;
#pragma unroll
        for (int e = 0; e < 16; ++e) As[r][c + e] = (half_t)src[e];
      }
    }
    // Stage B tile (32x128) f32 -> f16 LDS
    {
      int kr = tid >> 3;
      int c  = (tid & 7) * 16;
      const float* src = W + (size_t)(k0 + kr) * N + n0 + c;
#pragma unroll
      for (int e = 0; e < 16; ++e) Bs[kr][c + e] = (half_t)src[e];
    }
    __syncthreads();

    v16h a0 = lds_load_A(As, wm, lane);
    v16h a1 = lds_load_A(As, wm + 16, lane);
#pragma unroll
    for (int nt = 0; nt < 4; ++nt) {
      v16h b = lds_load_B(Bs, wn + nt * 16, lane);
      acc[0][nt] = wmma_f16(a0, b, acc[0][nt]);
      acc[1][nt] = wmma_f16(a1, b, acc[1][nt]);
    }
    __syncthreads();
  }

  // Epilogue. C/D layout: VGPR r -> M = r + 8*lh ; N = ln.
#pragma unroll
  for (int mt = 0; mt < 2; ++mt)
#pragma unroll
    for (int nt = 0; nt < 4; ++nt)
#pragma unroll
      for (int r = 0; r < 8; ++r) {
        int m = m0 + wm + mt * 16 + r + lh * 8;
        int n = n0 + wn + nt * 16 + ln;
        float v = acc[mt][nt][r];
        if (RELU) v = fmaxf(v, 0.f);
        size_t idx = (size_t)m * N + n;
        if (resid) v += resid[idx];
        if (outF32) outF32[idx] = v;
        if (outF16) outF16[idx] = (half_t)v;
      }
}

// ---------------------------------------------------------------------------
// Flash attention: one workgroup per (q-block of 128, head, batch).
// 8 waves; each wave owns 16 q-rows, full DH=64, online softmax over Sk=512.
// ---------------------------------------------------------------------------
__global__ __launch_bounds__(256) void attn_kernel(
    const half_t* __restrict__ Q, const half_t* __restrict__ Km,
    const half_t* __restrict__ Vm, const int* __restrict__ qlen,
    const int* __restrict__ klen, float* __restrict__ out) {
  const int qb  = blockIdx.x;   // 0..3
  const int h   = blockIdx.y;   // 0..7
  const int b   = blockIdx.z;   // 0..31
  const int tid = threadIdx.x;
  const int lane = tid & 31;
  const int w    = tid >> 5;
  const int ln   = lane & 15;
  const int lh   = lane >> 4;
  const int q0   = qb * 128 + w * 16;  // wave's first global q row
  const int kl   = klen[b];
  const int ql   = qlen[b];

  const half_t* Qb = Q  + (size_t)b * Sn * Dn + (size_t)h * DHn;
  const half_t* Kb = Km + (size_t)b * Sn * Dn + (size_t)h * DHn;
  const half_t* Vb = Vm + (size_t)b * Sn * Dn + (size_t)h * DHn;

  __shared__ half_t Pt[8][16][34];  // per-wave 16x32 P tile (transpose buffer)

  // Q A-operands for the two K-steps (DH = 64 = 2 x 32)
  v16h qa[2];
  {
    const half_t* qp = Qb + (size_t)(q0 + ln) * Dn;
#pragma unroll
    for (int ks = 0; ks < 2; ++ks) {
      int kh = ks * 32 + lh * 8;
#pragma unroll
      for (int e = 0; e < 8; ++e) {
        qa[ks][e]     = qp[kh + e];
        qa[ks][8 + e] = qp[kh + 16 + e];
      }
    }
  }

  float rmax[8], rsum[8];
#pragma unroll
  for (int r = 0; r < 8; ++r) { rmax[r] = -INFINITY; rsum[r] = 0.f; }
  v8f acc[4];
#pragma unroll
  for (int dt = 0; dt < 4; ++dt) {
    v8f z = {0.f, 0.f, 0.f, 0.f, 0.f, 0.f, 0.f, 0.f};
    acc[dt] = z;
  }

  const int qrow_base = q0 + lh * 8;

  for (int kb = 0; kb < Sn; kb += 32) {
    // ---- scores S (16q x 32k) = Q @ K^T / 8, with masks ----
    v8f sacc[2];
#pragma unroll
    for (int nt = 0; nt < 2; ++nt) {
      v8f z = {0.f, 0.f, 0.f, 0.f, 0.f, 0.f, 0.f, 0.f};
      sacc[nt] = z;
#pragma unroll
      for (int ks = 0; ks < 2; ++ks) {
        const half_t* kp = Kb + (size_t)(kb + nt * 16 + ln) * Dn + ks * 32 + lh * 8;
        v16h kB;
#pragma unroll
        for (int e = 0; e < 16; ++e) kB[e] = kp[e + (e >= 8 ? 8 : 0)];
        sacc[nt] = wmma_f16(qa[ks], kB, sacc[nt]);
      }
    }

    float sv[2][8];
    float tmax[8];
#pragma unroll
    for (int r = 0; r < 8; ++r) tmax[r] = -INFINITY;
#pragma unroll
    for (int nt = 0; nt < 2; ++nt) {
      int kk = kb + nt * 16 + ln;
      bool kvalid = (kk < kl);
#pragma unroll
      for (int r = 0; r < 8; ++r) {
        float s = sacc[nt][r] * 0.125f;
        int qq = qrow_base + r;
        if (!kvalid || kk == qq) s = NEGF;
        sv[nt][r] = s;
        tmax[r] = fmaxf(tmax[r], s);
      }
    }
    // row max across the 16 lanes of each half-group
#pragma unroll
    for (int m = 1; m <= 8; m <<= 1)
#pragma unroll
      for (int r = 0; r < 8; ++r)
        tmax[r] = fmaxf(tmax[r], __shfl_xor(tmax[r], m, 32));

    float scale[8];
#pragma unroll
    for (int r = 0; r < 8; ++r) {
      float nm = fmaxf(rmax[r], tmax[r]);
      scale[r] = __expf(rmax[r] - nm);  // first iter: exp(-inf) = 0
      rmax[r] = nm;
      rsum[r] *= scale[r];
    }
#pragma unroll
    for (int dt = 0; dt < 4; ++dt)
#pragma unroll
      for (int r = 0; r < 8; ++r) acc[dt][r] *= scale[r];

    // p = exp(s - rowmax); accumulate row sums; stage P into LDS transposed
    float radd[8];
#pragma unroll
    for (int r = 0; r < 8; ++r) radd[r] = 0.f;
#pragma unroll
    for (int nt = 0; nt < 2; ++nt)
#pragma unroll
      for (int r = 0; r < 8; ++r) {
        float e = __expf(sv[nt][r] - rmax[r]);
        radd[r] += e;
        Pt[w][lh * 8 + r][nt * 16 + ln] = (half_t)e;
      }
#pragma unroll
    for (int m = 1; m <= 8; m <<= 1)
#pragma unroll
      for (int r = 0; r < 8; ++r) radd[r] += __shfl_xor(radd[r], m, 32);
#pragma unroll
    for (int r = 0; r < 8; ++r) rsum[r] += radd[r];

    asm volatile("" ::: "memory");  // keep LDS store->load order (in-order per wave)

    // A-operand from P tile (A layout)
    v16h pa;
    {
      int kh = lh * 8;
#pragma unroll
      for (int e = 0; e < 8; ++e) {
        pa[e]     = Pt[w][ln][kh + e];
        pa[8 + e] = Pt[w][ln][16 + kh + e];
      }
    }
    // acc += P @ V  (4 d-tiles)
#pragma unroll
    for (int dt = 0; dt < 4; ++dt) {
      v16h vB;
      int d = dt * 16 + ln;
#pragma unroll
      for (int e = 0; e < 16; ++e)
        vB[e] = Vb[(size_t)(kb + lh * 8 + e + (e >= 8 ? 8 : 0)) * Dn + d];
      acc[dt] = wmma_f16(pa, vB, acc[dt]);
    }
  }

  // finalize: divide by rowsum, apply query mask, store f32
#pragma unroll
  for (int r = 0; r < 8; ++r) {
    int qq = qrow_base + r;
    float inv = 1.f / rsum[r];
    bool qvalid = (qq < ql);
#pragma unroll
    for (int dt = 0; dt < 4; ++dt) {
      float v = qvalid ? acc[dt][r] * inv : 0.f;
      out[((size_t)b * Sn + qq) * Dn + h * DHn + dt * 16 + ln] = v;
    }
  }
}

// ---------------------------------------------------------------------------
// LayerNorm over D=512 (one block per row). Optional residual add; writes
// f32 and optionally f16.
// ---------------------------------------------------------------------------
__global__ __launch_bounds__(256) void ln_kernel(
    const float* __restrict__ X, const float* __restrict__ R,
    const float* __restrict__ gamma, const float* __restrict__ beta,
    float* __restrict__ outF32, half_t* __restrict__ outF16) {
  const int row = blockIdx.x;
  const int tid = threadIdx.x;
  const size_t base = (size_t)row * Dn;
  float v0 = X[base + tid];
  float v1 = X[base + tid + 256];
  if (R) { v0 += R[base + tid]; v1 += R[base + tid + 256]; }

  __shared__ float red[256];
  red[tid] = v0 + v1;
  __syncthreads();
  for (int off = 128; off > 0; off >>= 1) {
    if (tid < off) red[tid] += red[tid + off];
    __syncthreads();
  }
  float mu = red[0] * (1.f / Dn);
  __syncthreads();
  float d0 = v0 - mu, d1 = v1 - mu;
  red[tid] = d0 * d0 + d1 * d1;
  __syncthreads();
  for (int off = 128; off > 0; off >>= 1) {
    if (tid < off) red[tid] += red[tid + off];
    __syncthreads();
  }
  float rs = rsqrtf(red[0] * (1.f / Dn) + LNEPS);
  float y0 = d0 * rs * gamma[tid] + beta[tid];
  float y1 = d1 * rs * gamma[tid + 256] + beta[tid + 256];
  if (outF32) { outF32[base + tid] = y0; outF32[base + tid + 256] = y1; }
  if (outF16) { outF16[base + tid] = (half_t)y0; outF16[base + tid + 256] = (half_t)y1; }
}

// mean over S: out[b,d] = mean_s X[b,s,d]
__global__ __launch_bounds__(256) void mean_kernel(const float* __restrict__ X,
                                                   float* __restrict__ out) {
  int d = blockIdx.x * 256 + threadIdx.x;
  int b = blockIdx.y;
  float s = 0.f;
  for (int t = 0; t < Sn; ++t) s += X[((size_t)b * Sn + t) * Dn + d];
  out[(size_t)b * Dn + d] = s * (1.f / Sn);
}

// ---------------------------------------------------------------------------
extern "C" void kernel_launch(void* const* d_in, const int* in_sizes, int n_in,
                              void* d_out, int out_size, void* d_ws, size_t ws_size,
                              hipStream_t stream) {
  const float* queries = (const float*)d_in[0];
  const float* keysp   = (const float*)d_in[1];
  const float* W_Q     = (const float*)d_in[2];
  const float* W_K     = (const float*)d_in[3];
  const float* W_V     = (const float*)d_in[4];
  const float* fw1     = (const float*)d_in[5];
  const float* fw2     = (const float*)d_in[6];
  const float* gamma   = (const float*)d_in[7];
  const float* beta    = (const float*)d_in[8];
  const int*   qlen    = (const int*)d_in[9];
  const int*   klen    = (const int*)d_in[10];

  char* ws = (char*)d_ws;
  const size_t MBy = 1ull << 20;
  half_t* Qh    = (half_t*)(ws + 0 * MBy);     // 16 MB
  half_t* Kh    = (half_t*)(ws + 16 * MBy);    // 16 MB
  half_t* Vh    = (half_t*)(ws + 32 * MBy);    // 16 MB
  float*  attnO = (float*)(ws + 48 * MBy);     // 32 MB
  float*  ln1f  = (float*)(ws + 80 * MBy);     // 32 MB
  half_t* ln1h  = (half_t*)(ws + 112 * MBy);   // 16 MB
  half_t* Hh    = (half_t*)(ws + 0 * MBy);     // 64 MB (reuses QKV+attn head)
  float*  y2    = (float*)(ws + 128 * MBy);    // 32 MB  -> peak 160 MB
  float*  ln2   = (float*)(ws + 0 * MBy);      // 32 MB (reuses Hh after FFN2)

  const int M = Bn * Sn;  // 16384
  dim3 blk(256);

  // QKV projections: [M,512] @ [512,512] -> f16
  gemm_wmma_kernel<false, false><<<dim3(Dn / 128, M / 128), blk, 0, stream>>>(
      queries, W_Q, nullptr, nullptr, Qh, M, Dn, Dn);
  gemm_wmma_kernel<false, false><<<dim3(Dn / 128, M / 128), blk, 0, stream>>>(
      keysp, W_K, nullptr, nullptr, Kh, M, Dn, Dn);
  gemm_wmma_kernel<false, false><<<dim3(Dn / 128, M / 128), blk, 0, stream>>>(
      keysp, W_V, nullptr, nullptr, Vh, M, Dn, Dn);

  // flash attention
  attn_kernel<<<dim3(Sn / 128, Hn, Bn), blk, 0, stream>>>(Qh, Kh, Vh, qlen, klen,
                                                          attnO);

  // result = LN(attn_out + queries)
  ln_kernel<<<dim3(M), blk, 0, stream>>>(attnO, queries, gamma, beta, ln1f, ln1h);

  // h = relu(result @ fw1)  [M,2048] f16
  gemm_wmma_kernel<true, true><<<dim3(DFFn / 128, M / 128), blk, 0, stream>>>(
      ln1h, fw1, nullptr, nullptr, Hh, M, DFFn, Dn);

  // y2 = result + h @ fw2
  gemm_wmma_kernel<true, false><<<dim3(Dn / 128, M / 128), blk, 0, stream>>>(
      Hh, fw2, ln1f, y2, nullptr, M, Dn, DFFn);

  // result = LN(y2)
  ln_kernel<<<dim3(M), blk, 0, stream>>>(y2, nullptr, gamma, beta, ln2, nullptr);

  // mean over S -> [B,1,D]
  mean_kernel<<<dim3(Dn / 256, Bn), blk, 0, stream>>>(ln2, (float*)d_out);
}
